// GNNModelGat_67740224193245
// MI455X (gfx1250) — matrix operand order repaired and checked by reading, not compiled, not run on hardware
//
#include <hip/hip_runtime.h>
#include <hip/hip_bf16.h>

// GAT + diffusion + MLP for MI455X (gfx1250), wave32.
// Needs ws_size >= ~71 MB (A matrix [4096x4096] f32 dominates).

#define NNODES 4096
#define FDIM   64
#define KC     32               // WMMA K per step (f16)
#define NEGINF (-9.0e15f)
#define LRELU(v) ((v) > 0.f ? (v) : 0.2f * (v))

typedef _Float16 v16h __attribute__((ext_vector_type(16)));
typedef float    v8f  __attribute__((ext_vector_type(8)));

union AF { v16h v; _Float16 h[16]; uint4 q[2]; };
union CF { v8f  v; float     f[8]; };
union P4 { _Float16 h[4]; uint2 u; };

// ---- CDNA5 async global->LDS copy path (ASYNCcnt), with safe fallback ----
#if __has_builtin(__builtin_amdgcn_global_load_async_to_lds_b128) && \
    __has_builtin(__builtin_amdgcn_s_wait_asynccnt)
#define GAT_ASYNC 1
typedef int v4i_t __attribute__((vector_size(16)));
typedef __attribute__((address_space(1))) v4i_t gv4i_t;
typedef __attribute__((address_space(3))) v4i_t lv4i_t;
#define ASYNC_CP16(dst, src) \
  __builtin_amdgcn_global_load_async_to_lds_b128((gv4i_t*)(src), (lv4i_t*)(dst), 0, 0)
#else
#define GAT_ASYNC 0
#endif

// ---------------- Wh1 = h @ W1 (f32, row-major; feeds s1/s2) ----------------
__global__ __launch_bounds__(256) void k_wh1(const float* __restrict__ h,
                                             const float* __restrict__ W1,
                                             float* __restrict__ Whf) {
  int gid = blockIdx.x * 256 + threadIdx.x;      // 4096*64
  int i = gid >> 6, c = gid & 63;
  float4 hv = *(const float4*)&h[i * 4];
  float acc = hv.x * W1[c] + hv.y * W1[64 + c] + hv.z * W1[128 + c] + hv.w * W1[192 + c];
  Whf[gid] = acc;
}

// ---------------- WhhT[c][i] = f16(Wh[i][c])  (coalesced transposed f16 copy) ------
__global__ __launch_bounds__(256) void k_whT(const float* __restrict__ h,
                                             const float* __restrict__ W1,
                                             _Float16* __restrict__ WhhT) {
  int gid = blockIdx.x * 256 + threadIdx.x;      // 64*4096, c-major
  int c = gid >> 12, i = gid & 4095;
  float4 hv = *(const float4*)&h[i * 4];
  float acc = hv.x * W1[c] + hv.y * W1[64 + c] + hv.z * W1[128 + c] + hv.w * W1[192 + c];
  WhhT[c * NNODES + i] = (_Float16)acc;
}

// ---------------- s1/s2 for layer 1: per-row dot-64 with a1 halves ----------------
__global__ __launch_bounds__(256) void k_s1(const float* __restrict__ Whf,
                                            const float* __restrict__ a1,
                                            float* __restrict__ s1,
                                            float* __restrict__ s2) {
  int i = blockIdx.x * 256 + threadIdx.x;
  float sa = 0.f, sb = 0.f;
#pragma unroll 8
  for (int c = 0; c < 64; ++c) {
    float w = Whf[i * 64 + c];
    sa += w * a1[c];
    sb += w * a1[64 + c];
  }
  s1[i] = sa;
  s2[i] = sb;
}

// ---------------- Fused GAT layer 1 (flash-softmax + WMMA accumulate) ----------------
// Block = 16 rows, 4 waves; wave w owns output columns [16w, 16w+16).
// B tiles ping-pong through LDS via async global->LDS loads (ASYNCcnt).
__global__ __launch_bounds__(128) void k_gat1_flash(const _Float16* __restrict__ WhT,
                                                    const float* __restrict__ s1,
                                                    const float* __restrict__ s2,
                                                    const float* __restrict__ adj,
                                                    float* __restrict__ xout) {
  __shared__ alignas(16) _Float16 sBT[2][FDIM][KC];  // transposed Wh tiles (ping-pong)
  __shared__ alignas(16) _Float16 sP[16][KC];        // exp(e - m) tile (A operand)
  __shared__ float sRedM[16][8];                     // partial row maxes
  __shared__ float sRedS[16][8];                     // partial row sums
  __shared__ float sM[16], sL[16], sScale[16];

  const int t     = threadIdx.x;
  const int lane  = t & 31;
  const int wave  = t >> 5;             // 0..3
  const int n0    = wave * 16;
  const int r0    = blockIdx.x * 16;
  const int rhalf = lane >> 4;          // 0/1
  const int rmod  = lane & 15;
  const int srow  = t >> 3;             // 0..15 (stats row owned by this thread)
  const int sq    = t & 7;              // 0..7  (4-col quarter)
  const int c0    = t >> 2;             // 0..31 (staging row), +32 for 2nd chunk
  const int part  = t & 3;              // 16B part within a 64B tile row

  if (t < 16) { sM[t] = NEGINF; sL[t] = 0.f; }
  CF acc;
#pragma unroll
  for (int v = 0; v < 8; ++v) acc.f[v] = 0.f;
  const float sArow = s1[r0 + srow];

  // stage a 64x32 f16 tile (4KB): each thread moves two 16B chunks
  auto stage = [&](int buf, int js) {
    const _Float16* g0 = WhT + c0 * NNODES + js + part * 8;
    const _Float16* g1 = WhT + (c0 + 32) * NNODES + js + part * 8;
    _Float16* d0 = &sBT[buf][c0][part * 8];
    _Float16* d1 = &sBT[buf][c0 + 32][part * 8];
#if GAT_ASYNC
    ASYNC_CP16(d0, g0);
    ASYNC_CP16(d1, g1);
#else
    *(uint4*)d0 = *(const uint4*)g0;
    *(uint4*)d1 = *(const uint4*)g1;
#endif
  };

  stage(0, 0);                          // prologue: tile 0 in flight
  __syncthreads();

  for (int j0 = 0; j0 < NNODES; j0 += KC) {
    const int cur = (j0 >> 5) & 1;
    // issue next tile into the other buffer (wrap keeps wait count constant)
    stage(cur ^ 1, (j0 + KC) & (NNODES - 1));

    // ---- logits: 4 per thread, kept in registers
    const int col0 = j0 + sq * 4;
    __builtin_prefetch(&adj[(r0 + srow) * NNODES + ((col0 + KC) & (NNODES - 1))], 0, 1);
    float4 av  = *(const float4*)&adj[(r0 + srow) * NNODES + col0];
    float4 s2v = *(const float4*)&s2[col0];
    float lg[4], lmax;
    {
      float v0 = sArow + s2v.x; v0 = LRELU(v0); if (av.x <= 0.f) v0 = NEGINF;
      float v1 = sArow + s2v.y; v1 = LRELU(v1); if (av.y <= 0.f) v1 = NEGINF;
      float v2 = sArow + s2v.z; v2 = LRELU(v2); if (av.z <= 0.f) v2 = NEGINF;
      float v3 = sArow + s2v.w; v3 = LRELU(v3); if (av.w <= 0.f) v3 = NEGINF;
      lg[0] = v0; lg[1] = v1; lg[2] = v2; lg[3] = v3;
      lmax = fmaxf(fmaxf(v0, v1), fmaxf(v2, v3));
    }
    sRedM[srow][sq] = lmax;
    __syncthreads();

    // ---- row max + rescale factor (16 threads)
    if (t < 16) {
      float mo = sM[t], cm = mo;
#pragma unroll
      for (int q = 0; q < 8; ++q) cm = fmaxf(cm, sRedM[t][q]);
      sScale[t] = __expf(mo - cm);
      sM[t] = cm;
    }
    __syncthreads();

    // ---- p = exp(e - m): 4 per thread; write f16 tile + partial sums
    {
      float mrow = sM[srow];
      P4 p4;
      float psum = 0.f;
#pragma unroll
      for (int q = 0; q < 4; ++q) {
        float p = __expf(lg[q] - mrow);
        p4.h[q] = (_Float16)p;
        psum += p;
      }
      *(uint2*)&sP[srow][sq * 4] = p4.u;
      sRedS[srow][sq] = psum;
    }
#if GAT_ASYNC
    // current tile (2 async ops) must be done; next tile (2 ops) may stay in flight
    __builtin_amdgcn_s_wait_asynccnt(2);
#endif
    __syncthreads();

    // ---- l update (16 lanes of wave 0; reconverges before WMMA)
    if (t < 16) {
      float s = 0.f;
#pragma unroll
      for (int q = 0; q < 8; ++q) s += sRedS[t][q];
      sL[t] = sL[t] * sScale[t] + s;
    }
    // ---- rescale accumulator, load fragments as b128, WMMA (EXEC all-ones)
#pragma unroll
    for (int v = 0; v < 8; ++v) acc.f[v] *= sScale[v + 8 * rhalf];

    AF a, b;
    {
      // A 16x32 f16: lane holds halves {rhalf*8+0..7, 16+rhalf*8+0..7} of row rmod
      const uint4* prow = (const uint4*)&sP[rmod][0];   // 4 x uint4 per row
      a.q[0] = prow[rhalf];
      a.q[1] = prow[2 + rhalf];
      // B 32x16 f16: lane holds halves {rhalf*8+0..7, 16+rhalf*8+0..7} of column n0+rmod
      const uint4* brow = (const uint4*)&sBT[cur][n0 + rmod][0];
      b.q[0] = brow[rhalf];
      b.q[1] = brow[2 + rhalf];
    }
    acc.v = __builtin_amdgcn_wmma_f32_16x16x32_f16(false, a.v, false, b.v,
                                                   (short)0, acc.v, false, false);
    __syncthreads();
  }

  // epilogue: normalize by l, apply ELU, store
#pragma unroll
  for (int v = 0; v < 8; ++v) {
    int row = v + 8 * rhalf;
    float val = acc.f[v] / sL[row];
    val = val > 0.f ? val : (__expf(val) - 1.f);
    xout[(r0 + row) * FDIM + n0 + rmod] = val;
  }
}

// ---------------- w2a = W2 @ a2 halves (layer-2 algebraic collapse) ----------------
__global__ void k_w2a(const float* __restrict__ W2, const float* __restrict__ a2,
                      float* __restrict__ w2a) {
  int k = threadIdx.x;  // 64 threads
  float sa = 0.f, sb = 0.f;
#pragma unroll 8
  for (int c = 0; c < 64; ++c) {
    float w = W2[k * 64 + c];
    sa += w * a2[c];
    sb += w * a2[64 + c];
  }
  w2a[k] = sa;
  w2a[64 + k] = sb;
}

// ---------------- s1_2/s2_2 = x1 @ w2a halves ----------------
__global__ __launch_bounds__(256) void k_s2(const float* __restrict__ x1,
                                            const float* __restrict__ w2a,
                                            float* __restrict__ s1,
                                            float* __restrict__ s2) {
  int i = blockIdx.x * 256 + threadIdx.x;
  float sa = 0.f, sb = 0.f;
#pragma unroll 8
  for (int k = 0; k < 64; ++k) {
    float x = x1[i * 64 + k];
    sa += x * w2a[k];
    sb += x * w2a[64 + k];
  }
  s1[i] = sa;
  s2[i] = sb;
}

// ---------------- layer-2 softmax row stats (one wave32 per row) ----------------
__global__ __launch_bounds__(256) void k_stats2(const float* __restrict__ s1,
                                                const float* __restrict__ s2,
                                                const float* __restrict__ adj,
                                                float* __restrict__ rmax,
                                                float* __restrict__ rsinv) {
  int j    = (blockIdx.x * 256 + threadIdx.x) >> 5;  // row
  int lane = threadIdx.x & 31;
  float sj = s1[j];
  float m = NEGINF, s = 0.f;
  for (int i = lane; i < NNODES; i += 32) {
    float v = sj + s2[i];
    v = LRELU(v);
    if (adj[j * NNODES + i] <= 0.f) v = NEGINF;
    if (v <= m) {
      s += __expf(v - m);
    } else {
      s = s * __expf(m - v) + 1.f;
      m = v;
    }
  }
  for (int off = 16; off; off >>= 1) {
    float om = __shfl_xor(m, off, 32);
    float os = __shfl_xor(s, off, 32);
    float nm = fmaxf(m, om);
    s = s * __expf(m - nm) + os * __expf(om - nm);
    m = nm;
  }
  if (lane == 0) { rmax[j] = m; rsinv[j] = 1.f / s; }
}

// ---------------- A[j,i] = prob[j,i] * att2[j,i]  (adj-layout; no transpose) --------
__global__ __launch_bounds__(256) void k_buildA(const float* __restrict__ s1,
                                                const float* __restrict__ s2,
                                                const float* __restrict__ adj,
                                                const float* __restrict__ prob,
                                                const float* __restrict__ rmax,
                                                const float* __restrict__ rsinv,
                                                float* __restrict__ A) {
  unsigned idx = blockIdx.x * 256 + threadIdx.x;  // < 2^24
  int j = idx >> 12, i = idx & 4095;
  float v = s1[j] + s2[i];
  v = LRELU(v);
  if (adj[idx] <= 0.f) v = NEGINF;
  float att = __expf(v - rmax[j]) * rsinv[j];
  A[idx] = prob[idx] * att;
}

// ---------------- seed into X[:,0] ----------------
__global__ void k_seed(const float* __restrict__ seed, float* __restrict__ X) {
  int i = blockIdx.x * 256 + threadIdx.x;
  X[i * 20] = seed[i];
}

// ---------------- diffusion step: X[:,k] via column reduction (A^T m) ----------------
__global__ __launch_bounds__(256) void k_diff(const float* __restrict__ A,
                                              float* __restrict__ X, int k) {
  __shared__ float sm[NNODES];                   // 16 KB broadcast vector
  for (int idx = threadIdx.x; idx < NNODES; idx += 256)
    sm[idx] = X[idx * 20 + (k - 1)];
  __syncthreads();

  int i = blockIdx.x * 256 + threadIdx.x;        // coalesced A columns across lanes
  float acc = 0.f;
#pragma unroll 8
  for (int j = 0; j < NNODES; ++j)
    acc += A[(size_t)j * NNODES + i] * sm[j];
  X[i * 20 + k] = acc;
}

// ---------------- tiny MLP head ----------------
__global__ __launch_bounds__(256) void k_mlp(const float* __restrict__ X,
                                             const float* __restrict__ w0,
                                             const float* __restrict__ b0,
                                             const float* __restrict__ w1,
                                             const float* __restrict__ b1,
                                             const float* __restrict__ w2,
                                             const float* __restrict__ b2,
                                             float* __restrict__ out) {
  __shared__ float sw0[20 * 64];
  __shared__ float sw1[64 * 64];
  __shared__ float sw2[64], sb0[64], sb1[64];
  for (int idx = threadIdx.x; idx < 20 * 64; idx += 256) sw0[idx] = w0[idx];
  for (int idx = threadIdx.x; idx < 64 * 64; idx += 256) sw1[idx] = w1[idx];
  if (threadIdx.x < 64) {
    sw2[threadIdx.x] = w2[threadIdx.x];
    sb0[threadIdx.x] = b0[threadIdx.x];
    sb1[threadIdx.x] = b1[threadIdx.x];
  }
  __syncthreads();

  int i = blockIdx.x * 256 + threadIdx.x;
  float xin[20];
#pragma unroll
  for (int k = 0; k < 20; ++k) xin[k] = X[i * 20 + k];
  float h0[64];
#pragma unroll 4
  for (int c = 0; c < 64; ++c) {
    float a = sb0[c];
#pragma unroll
    for (int k = 0; k < 20; ++k) a += xin[k] * sw0[k * 64 + c];
    h0[c] = fmaxf(a, 0.f);
  }
  float o = b2[0];
#pragma unroll 2
  for (int c = 0; c < 64; ++c) {
    float a = sb1[c];
#pragma unroll 8
    for (int k = 0; k < 64; ++k) a += h0[k] * sw1[k * 64 + c];
    o += fmaxf(a, 0.f) * sw2[c];
  }
  out[i] = 1.f / (1.f + __expf(-o));
}

extern "C" void kernel_launch(void* const* d_in, const int* in_sizes, int n_in,
                              void* d_out, int out_size, void* d_ws, size_t ws_size,
                              hipStream_t stream) {
  (void)in_sizes; (void)n_in; (void)out_size; (void)ws_size;
  const float* h    = (const float*)d_in[0];
  const float* seed = (const float*)d_in[1];
  const float* adj  = (const float*)d_in[2];
  const float* prob = (const float*)d_in[3];
  const float* W1   = (const float*)d_in[4];
  const float* a1   = (const float*)d_in[5];
  const float* W2   = (const float*)d_in[6];
  const float* a2   = (const float*)d_in[7];
  const float* fc0w = (const float*)d_in[8];
  const float* fc0b = (const float*)d_in[9];
  const float* fc1w = (const float*)d_in[10];
  const float* fc1b = (const float*)d_in[11];
  const float* fc2w = (const float*)d_in[12];
  const float* fc2b = (const float*)d_in[13];
  float* out = (float*)d_out;

  // workspace carve-up (floats)
  float* fws = (float*)d_ws;
  size_t o = 0;
  float*    Whf   = fws + o; o += (size_t)NNODES * FDIM;        // 1 MB
  _Float16* WhhT  = (_Float16*)(fws + o); o += (size_t)NNODES * FDIM / 2;  // 512 KB (f16, transposed)
  float*    s1_1  = fws + o; o += NNODES;
  float*    s2_1  = fws + o; o += NNODES;
  float*    x1    = fws + o; o += (size_t)NNODES * FDIM;        // 1 MB
  float*    w2a   = fws + o; o += 128;
  float*    s1_2  = fws + o; o += NNODES;
  float*    s2_2  = fws + o; o += NNODES;
  float*    rmax  = fws + o; o += NNODES;
  float*    rsinv = fws + o; o += NNODES;
  float*    X     = fws + o; o += (size_t)NNODES * 20;          // 320 KB
  float*    A     = fws + o; o += (size_t)NNODES * NNODES;      // 64 MB

  k_wh1<<<(NNODES * FDIM) / 256, 256, 0, stream>>>(h, W1, Whf);
  k_whT<<<(NNODES * FDIM) / 256, 256, 0, stream>>>(h, W1, WhhT);
  k_s1<<<NNODES / 256, 256, 0, stream>>>(Whf, a1, s1_1, s2_1);
  k_gat1_flash<<<NNODES / 16, 128, 0, stream>>>(WhhT, s1_1, s2_1, adj, x1);
  k_w2a<<<1, 64, 0, stream>>>(W2, a2, w2a);
  k_s2<<<NNODES / 256, 256, 0, stream>>>(x1, w2a, s1_2, s2_2);
  k_stats2<<<(NNODES * 32) / 256, 256, 0, stream>>>(s1_2, s2_2, adj, rmax, rsinv);
  k_buildA<<<(NNODES * NNODES) / 256, 256, 0, stream>>>(s1_2, s2_2, adj, prob, rmax, rsinv, A);
  k_seed<<<NNODES / 256, 256, 0, stream>>>(seed, X);
  for (int k = 1; k < 20; ++k)
    k_diff<<<NNODES / 256, 256, 0, stream>>>(A, X, k);
  k_mlp<<<NNODES / 256, 256, 0, stream>>>(X, fc0w, fc0b, fc1w, fc1b, fc2w, fc2b, out);
}
